// Conformer_9440338117412
// MI455X (gfx1250) — compile-verified, matrix operand order checked by reading, not run on hardware
//
#include <hip/hip_runtime.h>
#include <hip/hip_bf16.h>
#include <math.h>

// ---------------------------------------------------------------------------
// Conformer block for MI455X (gfx1250, wave32, WMMA).
// All GEMMs run on v_wmma_f32_16x16x32_bf16 (bf16 in, f32 accumulate).
// Attention is flash-style (online softmax) with the Transformer-XL rel_shift
// resolved through a per-block LDS table of (q+v)@pos^T rows.
// ---------------------------------------------------------------------------

typedef __attribute__((ext_vector_type(16))) __bf16 v16bf;
typedef __attribute__((ext_vector_type(8)))  float  v8f;

struct alignas(16) U128 { unsigned int x[4]; };

union FragU {
    U128 q[2];
    v16bf v;
    unsigned short s[16];
};

__device__ __forceinline__ unsigned short f2bf(float f) {
    unsigned int u = __float_as_uint(f);
    u += 0x7FFFu + ((u >> 16) & 1u);       // round-to-nearest-even
    return (unsigned short)(u >> 16);
}
__device__ __forceinline__ float bf2f(unsigned short h) {
    return __uint_as_float(((unsigned int)h) << 16);
}

__device__ __forceinline__ v8f wmma_bf16(v16bf a, v16bf b, v8f c) {
    return __builtin_amdgcn_wmma_f32_16x16x32_bf16(
        /*neg_a=*/false, a, /*neg_b=*/false, b,
        /*c_mod=*/(short)0, c, /*reuse_a=*/false, /*reuse_b=*/false);
}

// Fragment loader for row-major bf16 [rows][ld].  Per ISA 7.12.2 (16-bit A,
// and the mirrored B mapping): lane<16 -> row0+lane, K chunks {k0..k0+7,
// k0+16..k0+23}; lane>=16 -> row0+lane-16, K chunks {k0+8..15, k0+24..31}.
__device__ __forceinline__ v16bf load_frag(const unsigned short* __restrict__ base,
                                           int ld, int row0, int k0, int lane) {
    const int hh = lane >> 4, r = lane & 15;
    const unsigned short* p = base + (size_t)(row0 + r) * ld + k0 + hh * 8;
    FragU u;
    u.q[0] = *(const U128*)p;
    u.q[1] = *(const U128*)(p + 16);
    return u.v;
}

// ------------------------------ GEMM ---------------------------------------
// C[M,N] = epilogue( A[M,K](bf16) @ W[N,K]^T(bf16) ).  One wave = 32x32 tile.
// The time-shifted A path (k=3 conv over T) folds its per-row validity into
// per-lane base pointers BEFORE the K loop: invalid rows read a zero page with
// step 0, so the inner loop is pure b128 loads + pointer bumps + 4 wmma.
#define GF_ACC     1   // accumulate into existing C
#define GF_BIAS    2   // + bias[col]
#define GF_LRELU   4   // LeakyReLU slope 0.3
#define GF_HALFRES 8   // y = 0.5*y + res
#define GF_ADDRES  16  // y = y + res
#define GF_STOREBF 32  // also store bf16 copy

__global__ __launch_bounds__(256)
void gemm_kernel(const unsigned short* __restrict__ A, int lda,
                 const unsigned short* __restrict__ W,
                 const float* __restrict__ bias,
                 const float* __restrict__ res,
                 float* __restrict__ C, unsigned short* __restrict__ Cbf,
                 const unsigned short* __restrict__ zpage,
                 int M, int N, int K, int Tseq, int shift, int flags) {
    const int lane   = threadIdx.x & 31;
    const int wave   = blockIdx.x * 8 + (threadIdx.x >> 5);
    const int ntiles = N >> 5;
    const int wm = wave / ntiles, wn = wave - wm * ntiles;
    const int m0 = wm << 5, n0 = wn << 5;
    if (m0 >= M) return;

    const int hh = lane >> 4, r = lane & 15;

    // hoisted per-lane A row pointers (shift + sequence boundary folded in)
    const unsigned short* pa0;
    const unsigned short* pa1;
    int sa0, sa1;
    {
        const int row0 = m0 + r;
        const int row1 = m0 + 16 + r;
        const bool ok0 = (unsigned)((row0 % Tseq) + shift) < (unsigned)Tseq;
        const bool ok1 = (unsigned)((row1 % Tseq) + shift) < (unsigned)Tseq;
        pa0 = ok0 ? (A + (size_t)(row0 + shift) * lda + hh * 8) : zpage;
        pa1 = ok1 ? (A + (size_t)(row1 + shift) * lda + hh * 8) : zpage;
        sa0 = ok0 ? 32 : 0;
        sa1 = ok1 ? 32 : 0;
    }
    const unsigned short* pb0 = W + (size_t)(n0 + r) * K + hh * 8;
    const unsigned short* pb1 = W + (size_t)(n0 + 16 + r) * K + hh * 8;

    v8f acc00 = {}, acc01 = {}, acc10 = {}, acc11 = {};
    for (int k0 = 0; k0 < K; k0 += 32) {
        FragU a0, a1, b0, b1;
        a0.q[0] = *(const U128*)pa0;  a0.q[1] = *(const U128*)(pa0 + 16);
        a1.q[0] = *(const U128*)pa1;  a1.q[1] = *(const U128*)(pa1 + 16);
        b0.q[0] = *(const U128*)pb0;  b0.q[1] = *(const U128*)(pb0 + 16);
        b1.q[0] = *(const U128*)pb1;  b1.q[1] = *(const U128*)(pb1 + 16);
        pa0 += sa0; pa1 += sa1; pb0 += 32; pb1 += 32;
        acc00 = wmma_bf16(a0.v, b0.v, acc00);
        acc01 = wmma_bf16(a0.v, b1.v, acc01);
        acc10 = wmma_bf16(a1.v, b0.v, acc10);
        acc11 = wmma_bf16(a1.v, b1.v, acc11);
    }

    const int c16 = lane & 15;
    const int rb  = (lane >> 4) << 3;   // C layout: lane n -> col n%16, rows rb..rb+7
    v8f* accs[4] = { &acc00, &acc01, &acc10, &acc11 };
    #pragma unroll
    for (int ti = 0; ti < 2; ++ti) {
        #pragma unroll
        for (int tj = 0; tj < 2; ++tj) {
            v8f& a = *accs[ti * 2 + tj];
            const int col = n0 + tj * 16 + c16;
            const float bv = (flags & GF_BIAS) ? bias[col] : 0.f;
            #pragma unroll
            for (int rr = 0; rr < 8; ++rr) {
                const int row = m0 + ti * 16 + rb + rr;
                const size_t idx = (size_t)row * N + col;
                float y = a[rr];
                if (flags & GF_ACC)     y += C[idx];
                y += bv;
                if (flags & GF_LRELU)   y = (y >= 0.f) ? y : 0.3f * y;
                if (flags & GF_HALFRES) y = 0.5f * y + res[idx];
                if (flags & GF_ADDRES)  y += res[idx];
                C[idx] = y;
                if (flags & GF_STOREBF) Cbf[idx] = f2bf(y);
            }
        }
    }
}

// --------------------------- small kernels ---------------------------------
__global__ void zero_kernel(unsigned int* __restrict__ p, int n) {
    int i = blockIdx.x * blockDim.x + threadIdx.x;
    if (i < n) p[i] = 0u;
}

__global__ __launch_bounds__(128)
void ln_kernel(const float* __restrict__ x, const float* __restrict__ g,
               const float* __restrict__ bta, unsigned short* __restrict__ out, int D) {
    __shared__ float red[128];
    const int row = blockIdx.x;
    const float* xr = x + (size_t)row * D;
    float s = 0.f, ss = 0.f;
    for (int i = threadIdx.x; i < D; i += 128) { float v = xr[i]; s += v; ss += v * v; }
    red[threadIdx.x] = s; __syncthreads();
    for (int o = 64; o > 0; o >>= 1) { if (threadIdx.x < o) red[threadIdx.x] += red[threadIdx.x + o]; __syncthreads(); }
    const float mean = red[0] / D;
    __syncthreads();
    red[threadIdx.x] = ss; __syncthreads();
    for (int o = 64; o > 0; o >>= 1) { if (threadIdx.x < o) red[threadIdx.x] += red[threadIdx.x + o]; __syncthreads(); }
    const float var  = red[0] / D - mean * mean;
    const float rstd = rsqrtf(var + 1e-5f);
    unsigned short* orow = out + (size_t)row * D;
    for (int i = threadIdx.x; i < D; i += 128)
        orow[i] = f2bf((xr[i] - mean) * rstd * g[i] + bta[i]);
}

__global__ void cvt_bf16_kernel(const float* __restrict__ in, unsigned short* __restrict__ out, size_t n) {
    size_t i = (size_t)blockIdx.x * blockDim.x + threadIdx.x;
    if (i < n) out[i] = f2bf(in[i]);
}

__global__ void pack_w1_kernel(const float* __restrict__ w1, unsigned short* __restrict__ out) {
    int i = blockIdx.x * blockDim.x + threadIdx.x;           // 3 * 2048 * 512
    if (i >= 3 * 2048 * 512) return;
    const int plane = i / (2048 * 512);
    const int rem   = i - plane * (2048 * 512);              // o*512 + c
    out[i] = f2bf(w1[(size_t)rem * 3 + plane]);
}

__global__ void glu_kernel(const float* __restrict__ p, unsigned short* __restrict__ g, int n) {
    int i = blockIdx.x * blockDim.x + threadIdx.x;
    if (i >= n) return;
    const int m = i >> 10, c = i & 1023;
    const float a = p[(size_t)m * 2048 + c];
    float b = p[(size_t)m * 2048 + 1024 + c];
    b = (b >= 0.f) ? b : 0.3f * b;
    g[i] = f2bf(a * b);
}

__global__ void dwconv_kernel(const unsigned short* __restrict__ in, const float* __restrict__ w,
                              const float* __restrict__ bias, float* __restrict__ out, int n) {
    int i = blockIdx.x * blockDim.x + threadIdx.x;           // [4096][1024] over (b,t,c)
    if (i >= n) return;
    const int c = i & 1023, m = i >> 10, t = m & 1023;
    float s = bias[c];
    #pragma unroll
    for (int k = 0; k < 7; ++k) {
        const int ts = t + k - 3;
        if ((unsigned)ts < 1024u)
            s += bf2f(in[(size_t)(m + k - 3) * 1024 + c]) * w[c * 7 + k];
    }
    out[i] = s;
}

__global__ __launch_bounds__(1024)
void gn_reduce_kernel(const float* __restrict__ x, float* __restrict__ stats, int per) {
    __shared__ float r1[1024], r2[1024];
    const int b = blockIdx.x;
    const float* xb = x + (size_t)b * per;
    float s = 0.f, ss = 0.f;
    for (int i = threadIdx.x; i < per; i += 1024) { float v = xb[i]; s += v; ss += v * v; }
    r1[threadIdx.x] = s; r2[threadIdx.x] = ss; __syncthreads();
    for (int o = 512; o > 0; o >>= 1) {
        if (threadIdx.x < o) { r1[threadIdx.x] += r1[threadIdx.x + o]; r2[threadIdx.x] += r2[threadIdx.x + o]; }
        __syncthreads();
    }
    if (threadIdx.x == 0) {
        const float mean = r1[0] / per;
        const float var  = r2[0] / per - mean * mean;
        stats[b * 2]     = mean;
        stats[b * 2 + 1] = rsqrtf(var + 1e-5f);
    }
}

__global__ void gn_apply_kernel(const float* __restrict__ x, const float* __restrict__ stats,
                                const float* __restrict__ g, const float* __restrict__ bta,
                                unsigned short* __restrict__ out, int per, int n) {
    int i = blockIdx.x * blockDim.x + threadIdx.x;
    if (i >= n) return;
    const int b = i / per, c = i & 1023;
    float v = (x[i] - stats[b * 2]) * stats[b * 2 + 1];
    v = v * g[c] + bta[c];
    v = (v >= 0.f) ? v : 0.3f * v;
    out[i] = f2bf(v);
}

__global__ void maskz_kernel(float* __restrict__ x, const unsigned char* __restrict__ mask, int n) {
    int i = blockIdx.x * blockDim.x + threadIdx.x;
    if (i < n && mask[i >> 9]) x[i] = 0.f;
}

__global__ void prep_attn_kernel(const float* __restrict__ q, const float* __restrict__ k,
                                 const float* __restrict__ v, const float* __restrict__ p,
                                 const float* __restrict__ ub, const float* __restrict__ vb,
                                 unsigned short* __restrict__ qu, unsigned short* __restrict__ qv,
                                 unsigned short* __restrict__ kb, unsigned short* __restrict__ vt,
                                 unsigned short* __restrict__ pb) {
    int i = blockIdx.x * blockDim.x + threadIdx.x;           // b(2) h(4) t(10) d(5)
    if (i >= 4 * 16 * 1024 * 32) return;
    const int d = i & 31, t = (i >> 5) & 1023, h = (i >> 15) & 15, b = i >> 19;
    const int col = h * 32 + d;
    const size_t src = (size_t)(b * 1024 + t) * 512 + col;
    const float qq = q[src];
    qu[i] = f2bf(qq + ub[col]);
    qv[i] = f2bf(qq + vb[col]);
    kb[i] = f2bf(k[src]);
    vt[(size_t)((b * 16 + h) * 32 + d) * 1024 + t] = f2bf(v[src]);
    if (b == 0) pb[(size_t)(h * 1024 + t) * 32 + d] = f2bf(p[(size_t)t * 512 + col]);
}

// ------------------------- flash attention ----------------------------------
// One wave per (b, h, 16-query block).  LDS: PS rows t0..t0+16 (bf16) used to
// resolve rel_shift exactly:  shift[t,s] = PS[t, 1023+s-t] (s<=t), 0 (s==t+1),
// PS[t+1, s-t-2] (s>t+1).
__global__ __launch_bounds__(32)
void attn_kernel(const unsigned short* __restrict__ qu, const unsigned short* __restrict__ qv,
                 const unsigned short* __restrict__ kmat, const unsigned short* __restrict__ vt,
                 const unsigned short* __restrict__ posb, const unsigned char* __restrict__ mask,
                 unsigned short* __restrict__ ctx, float scale) {
    __shared__ unsigned short s_ps[17][1024];
    __shared__ float s_p[16][32];

    const int lane = threadIdx.x;
    const int tb = blockIdx.x & 63;
    const int h  = (blockIdx.x >> 6) & 15;
    const int b  = blockIdx.x >> 10;
    const int t0 = tb << 4;
    const size_t bh = (size_t)(b * 16 + h);
    const unsigned short* quh = qu   + bh * 1024 * 32;
    const unsigned short* qvh = qv   + bh * 1024 * 32;
    const unsigned short* kh  = kmat + bh * 1024 * 32;
    const unsigned short* vth = vt   + bh * 32 * 1024;
    const unsigned short* ph  = posb + (size_t)h * 1024 * 32;

    const v16bf fqu = load_frag(quh, 32, t0, 0, lane);
    const v16bf fqv = load_frag(qvh, 32, t0, 0, lane);

    const int c16 = lane & 15;
    const int rb  = (lane >> 4) << 3;

    // Phase A: PS rows t0..t0+15 via WMMA
    for (int c0 = 0; c0 < 1024; c0 += 16) {
        v16bf fp = load_frag(ph, 32, c0, 0, lane);
        v8f acc = {};
        acc = wmma_bf16(fqv, fp, acc);
        const int col = c0 + c16;
        #pragma unroll
        for (int r = 0; r < 8; ++r) s_ps[rb + r][col] = f2bf(acc[r]);
    }
    // Extra PS row t0+16 (VALU dot products)
    {
        const int t16 = t0 + 16;
        if (t16 < 1024) {
            float q16[32];
            const unsigned short* qrow = qvh + (size_t)t16 * 32;
            #pragma unroll
            for (int d = 0; d < 32; ++d) q16[d] = bf2f(qrow[d]);
            for (int c = lane; c < 1024; c += 32) {
                const unsigned short* pr = ph + (size_t)c * 32;
                float s = 0.f;
                #pragma unroll
                for (int d = 0; d < 32; ++d) s += q16[d] * bf2f(pr[d]);
                s_ps[16][c] = f2bf(s);
            }
        } else {
            for (int c = lane; c < 1024; c += 32) s_ps[16][c] = 0;
        }
    }
    __syncthreads();

    float m8[8], l8[8];
    #pragma unroll
    for (int r = 0; r < 8; ++r) { m8[r] = -3.0e38f; l8[r] = 0.f; }
    v8f c0acc = {}, c1acc = {};
    const unsigned char* mrow = mask + (size_t)b * 1024;

    for (int s0 = 0; s0 < 1024; s0 += 32) {
        // content scores for two 16-wide key tiles (K = d_head = 32, one WMMA each)
        v8f sc[2];
        #pragma unroll
        for (int j = 0; j < 2; ++j) {
            v16bf fk = load_frag(kh, 32, s0 + j * 16, 0, lane);
            v8f z = {};
            sc[j] = wmma_bf16(fqu, fk, z);
        }
        // add shifted positional score, scale, mask
        #pragma unroll
        for (int j = 0; j < 2; ++j) {
            const int s = s0 + j * 16 + c16;
            const bool mk = mrow[s] != 0;
            #pragma unroll
            for (int r = 0; r < 8; ++r) {
                const int t = t0 + rb + r;
                float pv;
                if (s <= t)          pv = bf2f(s_ps[rb + r][s - t + 1023]);
                else if (s == t + 1) pv = 0.f;
                else                 pv = bf2f(s_ps[rb + r + 1][s - t - 2]);
                const float val = (sc[j][r] + pv) * scale;
                sc[j][r] = mk ? -1e9f : val;
            }
        }
        // online softmax update (rows live in 16-lane halves)
        float alpha[8];
        #pragma unroll
        for (int r = 0; r < 8; ++r) {
            float mx = fmaxf(sc[0][r], sc[1][r]);
            #pragma unroll
            for (int off = 8; off > 0; off >>= 1) mx = fmaxf(mx, __shfl_xor(mx, off, 16));
            const float mnew = fmaxf(m8[r], mx);
            const float a = __expf(m8[r] - mnew);
            float ps = 0.f;
            #pragma unroll
            for (int j = 0; j < 2; ++j) {
                const float e = __expf(sc[j][r] - mnew);
                sc[j][r] = e; ps += e;
            }
            #pragma unroll
            for (int off = 8; off > 0; off >>= 1) ps += __shfl_xor(ps, off, 16);
            l8[r] = l8[r] * a + ps;
            m8[r] = mnew;
            alpha[r] = a;
        }
        #pragma unroll
        for (int r = 0; r < 8; ++r) { c0acc[r] *= alpha[r]; c1acc[r] *= alpha[r]; }

        // stage probs (C layout) -> LDS -> rebuild as A fragment (bf16)
        #pragma unroll
        for (int j = 0; j < 2; ++j)
            #pragma unroll
            for (int r = 0; r < 8; ++r) s_p[rb + r][j * 16 + c16] = sc[j][r];
        __syncthreads();

        FragU ua;
        {
            const int ar = lane & 15, ah = lane >> 4;
            #pragma unroll
            for (int i2 = 0; i2 < 8; ++i2) {
                ua.s[i2]     = f2bf(s_p[ar][ah * 8 + i2]);
                ua.s[i2 + 8] = f2bf(s_p[ar][ah * 8 + 16 + i2]);
            }
        }
        const v16bf fa  = ua.v;
        const v16bf fv0 = load_frag(vth, 1024, 0,  s0, lane);  // V^T rows = d
        const v16bf fv1 = load_frag(vth, 1024, 16, s0, lane);
        c0acc = wmma_bf16(fa, fv0, c0acc);
        c1acc = wmma_bf16(fa, fv1, c1acc);
        __syncthreads();
    }

    // normalize and store context (bf16, [b,t,h*32+d] for the Wo GEMM)
    unsigned short* op = ctx + (size_t)(b * 1024 + t0) * 512 + h * 32;
    #pragma unroll
    for (int r = 0; r < 8; ++r) {
        const float inv = 1.f / l8[r];
        op[(size_t)(rb + r) * 512 + c16]      = f2bf(c0acc[r] * inv);
        op[(size_t)(rb + r) * 512 + 16 + c16] = f2bf(c1acc[r] * inv);
    }
}

// ------------------------------ host side ----------------------------------
extern "C" void kernel_launch(void* const* d_in, const int* in_sizes, int n_in,
                              void* d_out, int out_size, void* d_ws, size_t ws_size,
                              hipStream_t stream) {
    (void)in_sizes; (void)n_in; (void)out_size; (void)ws_size;

    const int Mrows = 4096;           // B*T
    // Input order follows setup_inputs(): x, mask, encoding, then params
    // flattened depth-first in dict order.
    const float* x_in  = (const float*)d_in[0];
    const unsigned char* mask = (const unsigned char*)d_in[1];
    const float* enc   = (const float*)d_in[2];
    const float* ff_ln_g = (const float*)d_in[3];
    const float* ff_ln_b = (const float*)d_in[4];
    const float* ff_w1   = (const float*)d_in[5];
    const float* ff_b1   = (const float*)d_in[6];
    const float* ff_w2   = (const float*)d_in[7];
    const float* ff_b2   = (const float*)d_in[8];
    // cc1: 9..18 ; cc2: 19..28
    const float* ln_g  = (const float*)d_in[29];
    const float* ln_b  = (const float*)d_in[30];
    const float* at_wq = (const float*)d_in[31];
    const float* at_bq = (const float*)d_in[32];
    const float* at_wk = (const float*)d_in[33];
    const float* at_wv = (const float*)d_in[34];
    const float* at_wp = (const float*)d_in[35];
    const float* at_u  = (const float*)d_in[36];
    const float* at_v  = (const float*)d_in[37];
    const float* at_wo = (const float*)d_in[38];
    const float* at_bo = (const float*)d_in[39];

    float* xcur = (float*)d_out;      // running activation [4096, 512]

    // workspace allocator (256B aligned)
    char* base = (char*)d_ws;
    size_t off = 0;
    auto alloc = [&](size_t bytes) -> void* {
        void* p = base + off;
        off += (bytes + 255) & ~(size_t)255;
        return p;
    };
    unsigned short* zpage = (unsigned short*)alloc(512);   // zero page for OOB conv rows
    unsigned short* abf   = (unsigned short*)alloc((size_t)Mrows * 512 * 2);
    unsigned short* hbf   = (unsigned short*)alloc((size_t)Mrows * 2048 * 2);
    float*          bigf  = (float*)        alloc((size_t)Mrows * 2048 * 4);
    unsigned short* gbf   = (unsigned short*)alloc((size_t)Mrows * 1024 * 2);
    unsigned short* nbf   = (unsigned short*)alloc((size_t)Mrows * 1024 * 2);
    float*          stats = (float*)        alloc(64 * 4);
    unsigned short* w1bf  = (unsigned short*)alloc((size_t)3 * 2048 * 512 * 2);
    unsigned short* w2bf  = (unsigned short*)alloc((size_t)512 * 2048 * 2);
    unsigned short* pw1bf1= (unsigned short*)alloc((size_t)2048 * 512 * 2);
    unsigned short* pw1bf2= (unsigned short*)alloc((size_t)2048 * 512 * 2);
    unsigned short* pw2bf1= (unsigned short*)alloc((size_t)512 * 1024 * 2);
    unsigned short* pw2bf2= (unsigned short*)alloc((size_t)512 * 1024 * 2);
    unsigned short* wqbf  = (unsigned short*)alloc((size_t)512 * 512 * 2);
    unsigned short* wkbf  = (unsigned short*)alloc((size_t)512 * 512 * 2);
    unsigned short* wvbf  = (unsigned short*)alloc((size_t)512 * 512 * 2);
    unsigned short* wpbf  = (unsigned short*)alloc((size_t)512 * 512 * 2);
    unsigned short* wobf  = (unsigned short*)alloc((size_t)512 * 512 * 2);
    unsigned short* encbf = (unsigned short*)alloc((size_t)1024 * 512 * 2);
    unsigned short* qubf  = (unsigned short*)alloc((size_t)2097152 * 2);
    unsigned short* qvbf  = (unsigned short*)alloc((size_t)2097152 * 2);
    unsigned short* kbbf  = (unsigned short*)alloc((size_t)2097152 * 2);
    unsigned short* vtbf  = (unsigned short*)alloc((size_t)2097152 * 2);
    unsigned short* posbf = (unsigned short*)alloc((size_t)16 * 1024 * 32 * 2);
    unsigned short* ctxbf = (unsigned short*)alloc((size_t)Mrows * 512 * 2);

    auto cvt = [&](const float* in, unsigned short* o, size_t n) {
        cvt_bf16_kernel<<<(unsigned)((n + 255) / 256), 256, 0, stream>>>(in, o, n);
    };
    auto gemm = [&](const unsigned short* A, int lda, const unsigned short* W,
                    const float* bias, const float* res, float* C, unsigned short* Cbf,
                    int M, int N, int K, int shift, int flags) {
        const int waves  = (M / 32) * (N / 32);
        const int blocks = (waves + 7) / 8;
        gemm_kernel<<<blocks, 256, 0, stream>>>(A, lda, W, bias, res, C, Cbf, zpage,
                                                M, N, K, 1024, shift, flags);
    };

    // zero page + x -> d_out (running buffer)
    zero_kernel<<<1, 128, 0, stream>>>((unsigned int*)zpage, 128);
    hipMemcpyAsync(xcur, x_in, sizeof(float) * (size_t)Mrows * 512,
                   hipMemcpyDeviceToDevice, stream);

    // ---- weight prep (fp32 -> bf16, pack conv3 planes) ----
    pack_w1_kernel<<<(3 * 2048 * 512 + 255) / 256, 256, 0, stream>>>(ff_w1, w1bf);
    cvt(ff_w2, w2bf, (size_t)512 * 2048);
    cvt((const float*)d_in[11], pw1bf1, (size_t)2048 * 512);
    cvt((const float*)d_in[17], pw2bf1, (size_t)512 * 1024);
    cvt((const float*)d_in[21], pw1bf2, (size_t)2048 * 512);
    cvt((const float*)d_in[27], pw2bf2, (size_t)512 * 1024);
    cvt(at_wq, wqbf, (size_t)512 * 512);
    cvt(at_wk, wkbf, (size_t)512 * 512);
    cvt(at_wv, wvbf, (size_t)512 * 512);
    cvt(at_wp, wpbf, (size_t)512 * 512);
    cvt(at_wo, wobf, (size_t)512 * 512);
    cvt(enc, encbf, (size_t)1024 * 512);

    // ---- feed-forward half-step: LN -> conv(k=3) -> lrelu -> conv(k=1) -> 0.5 + res ----
    ln_kernel<<<Mrows, 128, 0, stream>>>(xcur, ff_ln_g, ff_ln_b, abf, 512);
    gemm(abf, 512, w1bf + 0 * (size_t)2048 * 512, nullptr, nullptr, bigf, hbf,
         Mrows, 2048, 512, -1, 0);
    gemm(abf, 512, w1bf + 1 * (size_t)2048 * 512, nullptr, nullptr, bigf, hbf,
         Mrows, 2048, 512, 0, GF_ACC);
    gemm(abf, 512, w1bf + 2 * (size_t)2048 * 512, ff_b1, nullptr, bigf, hbf,
         Mrows, 2048, 512, 1, GF_ACC | GF_BIAS | GF_LRELU | GF_STOREBF);
    gemm(hbf, 2048, w2bf, ff_b2, xcur, xcur, nullptr,
         Mrows, 512, 2048, 0, GF_BIAS | GF_HALFRES);

    // ---- conv module ----
    auto run_cc = [&](const float* lng, const float* lnb,
                      unsigned short* pw1w, const float* pw1b,
                      const float* dww, const float* dwb,
                      const float* gng, const float* gnb,
                      unsigned short* pw2w, const float* pw2b) {
        ln_kernel<<<Mrows, 128, 0, stream>>>(xcur, lng, lnb, abf, 512);
        gemm(abf, 512, pw1w, pw1b, nullptr, bigf, nullptr,
             Mrows, 2048, 512, 0, GF_BIAS);
        glu_kernel<<<(Mrows * 1024 + 255) / 256, 256, 0, stream>>>(bigf, gbf, Mrows * 1024);
        dwconv_kernel<<<(Mrows * 1024 + 255) / 256, 256, 0, stream>>>(gbf, dww, dwb, bigf, Mrows * 1024);
        gn_reduce_kernel<<<4, 1024, 0, stream>>>(bigf, stats, 1024 * 1024);
        gn_apply_kernel<<<(Mrows * 1024 + 255) / 256, 256, 0, stream>>>(
            bigf, stats, gng, gnb, nbf, 1024 * 1024, Mrows * 1024);
        gemm(nbf, 1024, pw2w, pw2b, xcur, xcur, nullptr,
             Mrows, 512, 1024, 0, GF_BIAS | GF_ADDRES);
    };

    run_cc((const float*)d_in[9],  (const float*)d_in[10], pw1bf1, (const float*)d_in[12],
           (const float*)d_in[13], (const float*)d_in[14], (const float*)d_in[15],
           (const float*)d_in[16], pw2bf1, (const float*)d_in[18]);

    // ---- relative-position MHSA ----
    ln_kernel<<<Mrows, 128, 0, stream>>>(xcur, ln_g, ln_b, abf, 512);
    float* qf = bigf;
    float* kf = bigf + (size_t)Mrows * 512;
    float* vf = bigf + (size_t)2 * Mrows * 512;
    float* pf = bigf + (size_t)3 * Mrows * 512;
    gemm(abf, 512, wqbf, at_bq, nullptr, qf, nullptr, Mrows, 512, 512, 0, GF_BIAS);
    gemm(abf, 512, wkbf, nullptr, nullptr, kf, nullptr, Mrows, 512, 512, 0, 0);
    gemm(abf, 512, wvbf, nullptr, nullptr, vf, nullptr, Mrows, 512, 512, 0, 0);
    gemm(encbf, 512, wpbf, nullptr, nullptr, pf, nullptr, 1024, 512, 512, 0, 0);
    prep_attn_kernel<<<(2097152 + 255) / 256, 256, 0, stream>>>(
        qf, kf, vf, pf, at_u, at_v, qubf, qvbf, kbbf, vtbf, posbf);
    attn_kernel<<<4096, 32, 0, stream>>>(qubf, qvbf, kbbf, vtbf, posbf, mask,
                                         ctxbf, 1.0f / sqrtf(512.0f));
    gemm(ctxbf, 512, wobf, at_bo, xcur, xcur, nullptr,
         Mrows, 512, 512, 0, GF_BIAS | GF_ADDRES);
    maskz_kernel<<<(Mrows * 512 + 255) / 256, 256, 0, stream>>>(xcur, mask, Mrows * 512);

    // ---- second conv module ----
    run_cc((const float*)d_in[19], (const float*)d_in[20], pw1bf2, (const float*)d_in[22],
           (const float*)d_in[23], (const float*)d_in[24], (const float*)d_in[25],
           (const float*)d_in[26], pw2bf2, (const float*)d_in[28]);
}
